// LSTMCell_32779190403217
// MI455X (gfx1250) — matrix operand order repaired and checked by reading, not compiled, not run on hardware
//
#include <hip/hip_runtime.h>

// LSTM cell fused as one bf16 WMMA GEMM:
//   gates[4096 x 4096] = A[4096 x 2048] * Bt^T, columns interleaved n = 4*h + g
//   gate order g: 0=f, 1=i, 2=o, 3=c
// Async (ASYNCcnt) double-buffered LDS staging, branch-free pipelined K loop,
// fused sigmoid/tanh LSTM epilogue.

typedef __attribute__((ext_vector_type(16))) __bf16     v16bf;
typedef __attribute__((ext_vector_type(8)))  float       v8f;
typedef __attribute__((ext_vector_type(4)))  unsigned int v4u;

#define BATCH   4096
#define HID     1024
#define KDIM    2048   // IN + HIDDEN
#define NDIM    4096   // 4 gates * HIDDEN, interleaved
#define BM      128
#define BN      128
#define BK      32
#define LDSS    40     // padded LDS row stride (bf16 elems): 80B rows, conflict-free frags
#define CSS     68     // padded epilogue stride (f32 elems): 272B rows, 16B-aligned quads

__device__ __forceinline__ unsigned short f2bf(float f) {
  unsigned int u = __float_as_uint(f);
  unsigned int r = u + 0x7FFFu + ((u >> 16) & 1u);   // round-to-nearest-even
  return (unsigned short)(r >> 16);
}

// ---------------- pack A: [x | hidden] -> bf16 A[4096][2048] ----------------
__global__ __launch_bounds__(256) void pack_a_kernel(const float* __restrict__ x,
                                                     const float* __restrict__ hid,
                                                     unsigned short* __restrict__ Abf) {
  size_t e = ((size_t)blockIdx.x * 256 + threadIdx.x) * 8;
  int m = (int)(e >> 11);
  int k = (int)(e & 2047);
  const float* src = (k < 1024) ? (x + (size_t)m * 1024 + k)
                                : (hid + (size_t)m * 1024 + (k - 1024));
  float4 f0 = *(const float4*)src;
  float4 f1 = *(const float4*)(src + 4);
  union { v4u q; unsigned short h[8]; } o;
  o.h[0] = f2bf(f0.x); o.h[1] = f2bf(f0.y); o.h[2] = f2bf(f0.z); o.h[3] = f2bf(f0.w);
  o.h[4] = f2bf(f1.x); o.h[5] = f2bf(f1.y); o.h[6] = f2bf(f1.z); o.h[7] = f2bf(f1.w);
  *(v4u*)(Abf + e) = o.q;
}

// ---- pack B: Bt[n][k] = concat(W_x[g][h][:], W_h[g][h][:]), n = 4h+g ------
__global__ __launch_bounds__(256) void pack_b_kernel(
    const float* __restrict__ Wx0, const float* __restrict__ Wx1,
    const float* __restrict__ Wx2, const float* __restrict__ Wx3,
    const float* __restrict__ Wh0, const float* __restrict__ Wh1,
    const float* __restrict__ Wh2, const float* __restrict__ Wh3,
    const float* __restrict__ bx0, const float* __restrict__ bx1,
    const float* __restrict__ bx2, const float* __restrict__ bx3,
    const float* __restrict__ bh0, const float* __restrict__ bh1,
    const float* __restrict__ bh2, const float* __restrict__ bh3,
    unsigned short* __restrict__ Bbf, float* __restrict__ bias) {
  int n = blockIdx.x;
  int g = n & 3;
  int h = n >> 2;
  const float* wx = (g == 0) ? Wx0 : (g == 1) ? Wx1 : (g == 2) ? Wx2 : Wx3;
  const float* wh = (g == 0) ? Wh0 : (g == 1) ? Wh1 : (g == 2) ? Wh2 : Wh3;
  int k = threadIdx.x * 8;
  const float* src = (k < 1024) ? (wx + (size_t)h * 1024 + k)
                                : (wh + (size_t)h * 1024 + (k - 1024));
  float4 f0 = *(const float4*)src;
  float4 f1 = *(const float4*)(src + 4);
  union { v4u q; unsigned short hh[8]; } o;
  o.hh[0] = f2bf(f0.x); o.hh[1] = f2bf(f0.y); o.hh[2] = f2bf(f0.z); o.hh[3] = f2bf(f0.w);
  o.hh[4] = f2bf(f1.x); o.hh[5] = f2bf(f1.y); o.hh[6] = f2bf(f1.z); o.hh[7] = f2bf(f1.w);
  *(v4u*)(Bbf + (size_t)n * KDIM + k) = o.q;
  if (threadIdx.x == 0) {
    const float* bx = (g == 0) ? bx0 : (g == 1) ? bx1 : (g == 2) ? bx2 : bx3;
    const float* bh = (g == 0) ? bh0 : (g == 1) ? bh1 : (g == 2) ? bh2 : bh3;
    bias[n] = bx[h] + bh[h];
  }
}

// ---------------- fused GEMM + LSTM epilogue ----------------
// Grid (NDIM/BN, BATCH/BM); 128 threads = 4 waves, 64x64 per wave.
__global__ __launch_bounds__(128) void lstm_gemm_kernel(
    const unsigned short* __restrict__ Abf,   // [BATCH][KDIM] bf16
    const unsigned short* __restrict__ Bbf,   // [NDIM][KDIM]  bf16 (n-major)
    const float* __restrict__ bias,           // [NDIM]
    const float* __restrict__ cell,           // [BATCH][HID]
    float* __restrict__ out_h,                // [BATCH][HID]
    float* __restrict__ out_c) {              // [BATCH][HID]
  __shared__ __align__(16) unsigned short As[2][BM][LDSS];   // 2 x 10240 B
  __shared__ __align__(16) unsigned short Bs[2][BN][LDSS];   // 2 x 10240 B
  __shared__ __align__(16) float Cs[4][16][CSS];             // 17408 B

  const int t    = threadIdx.x;
  const int w    = t >> 5;          // wave 0..3
  const int wm   = w >> 1;          // wave row in 2x2 grid
  const int wn   = w & 1;           // wave col
  const int lane = t & 31;
  const int lrow = lane & 15;
  const int lhi  = lane >> 4;

  const int m0 = blockIdx.y * BM;
  const int n0 = blockIdx.x * BN;

  // per-thread byte offsets of this thread's staging row in A / B
  const unsigned aoff0 = (unsigned)(m0 + t) * (KDIM * 2);
  const unsigned boff0 = (unsigned)(n0 + t) * (KDIM * 2);
  // per-thread LDS byte addresses of staging row (low 32 bits of flat ptr = LDS addr)
  const unsigned ldsA0 = (unsigned)(size_t)&As[0][t][0];
  const unsigned ldsA1 = (unsigned)(size_t)&As[1][t][0];
  const unsigned ldsB0 = (unsigned)(size_t)&Bs[0][t][0];
  const unsigned ldsB1 = (unsigned)(size_t)&Bs[1][t][0];

  // Async copy of one 64B row-slice each of A and B into LDS (8 wave-instrs / tile).
  // IOFFSET applies to both the global and the LDS address (ISA 10.x async pseudocode).
#define ASYNC_TILE(la, lb, av, bv)                                          \
  asm volatile(                                                             \
      "global_load_async_to_lds_b128 %0, %2, %4 offset:0\n\t"               \
      "global_load_async_to_lds_b128 %0, %2, %4 offset:16\n\t"              \
      "global_load_async_to_lds_b128 %0, %2, %4 offset:32\n\t"              \
      "global_load_async_to_lds_b128 %0, %2, %4 offset:48\n\t"              \
      "global_load_async_to_lds_b128 %1, %3, %5 offset:0\n\t"               \
      "global_load_async_to_lds_b128 %1, %3, %5 offset:16\n\t"              \
      "global_load_async_to_lds_b128 %1, %3, %5 offset:32\n\t"              \
      "global_load_async_to_lds_b128 %1, %3, %5 offset:48"                  \
      :: "v"(la), "v"(lb), "v"(av), "v"(bv), "s"(Abf), "s"(Bbf) : "memory")

#define WAIT_ASYNC8() asm volatile("s_wait_asynccnt 8" ::: "memory")
#define WAIT_ASYNC0() asm volatile("s_wait_asynccnt 0" ::: "memory")

#define COMPUTE_TILE(BUF)                                                     \
  do {                                                                        \
    union Frag { v16bf v; v4u q[2]; };                                        \
    Frag fa[4], fb[4];                                                        \
    _Pragma("unroll")                                                         \
    for (int mi = 0; mi < 4; ++mi) {                                          \
      const unsigned short* ap = &As[BUF][wm * 64 + mi * 16 + lrow][lhi * 8]; \
      fa[mi].q[0] = *(const v4u*)ap;                                          \
      fa[mi].q[1] = *(const v4u*)(ap + 16);                                   \
    }                                                                         \
    _Pragma("unroll")                                                         \
    for (int ni = 0; ni < 4; ++ni) {                                          \
      const unsigned short* bp = &Bs[BUF][wn * 64 + ni * 16 + lrow][lhi * 8]; \
      fb[ni].q[0] = *(const v4u*)bp;                                          \
      fb[ni].q[1] = *(const v4u*)(bp + 16);                                   \
    }                                                                         \
    _Pragma("unroll")                                                         \
    for (int mi = 0; mi < 4; ++mi)                                            \
      _Pragma("unroll")                                                       \
      for (int ni = 0; ni < 4; ++ni)                                          \
        acc[mi][ni] = __builtin_amdgcn_wmma_f32_16x16x32_bf16(                \
            false, fa[mi].v, false, fb[ni].v, (short)0, acc[mi][ni],          \
            false, false);                                                    \
  } while (0)

  v8f acc[4][4] = {};

  ASYNC_TILE(ldsA0, ldsB0, aoff0, boff0);              // tile 0 -> buf0

  int kt = 0;
  // steady state: two tiles per iteration, fully branch-free bodies
  for (; kt < KDIM - 2 * BK; kt += 2 * BK) {
    ASYNC_TILE(ldsA1, ldsB1, aoff0 + (kt + BK) * 2, boff0 + (kt + BK) * 2);
    WAIT_ASYNC8();                                     // tile kt resident (in-order)
    __syncthreads();
    COMPUTE_TILE(0);
    __syncthreads();                                   // buf0 free for reuse
    ASYNC_TILE(ldsA0, ldsB0, aoff0 + (kt + 2 * BK) * 2, boff0 + (kt + 2 * BK) * 2);
    WAIT_ASYNC8();                                     // tile kt+BK resident
    __syncthreads();
    COMPUTE_TILE(1);
    __syncthreads();                                   // buf1 free for reuse
  }
  // tail: tiles kt (buf0, already prefetched) and kt+BK (buf1)
  ASYNC_TILE(ldsA1, ldsB1, aoff0 + (kt + BK) * 2, boff0 + (kt + BK) * 2);
  WAIT_ASYNC8();
  __syncthreads();
  COMPUTE_TILE(0);
  __syncthreads();
  WAIT_ASYNC0();                                       // last tile resident
  __syncthreads();
  COMPUTE_TILE(1);

#undef ASYNC_TILE
#undef WAIT_ASYNC8
#undef WAIT_ASYNC0
#undef COMPUTE_TILE

  // ---------------- epilogue: four 16-row chunks per wave ----------------
  const int nbase = n0 + wn * 64;          // this wave's first gate column
  const int hg0   = nbase >> 2;            // first hidden index
  const int h     = lane & 15;
  const int r0    = lane >> 4;
  float4 b4 = *(const float4*)(bias + nbase + h * 4);
#pragma unroll
  for (int ch = 0; ch < 4; ++ch) {
    __syncthreads();
#pragma unroll
    for (int ni = 0; ni < 4; ++ni)
#pragma unroll
      for (int r = 0; r < 8; ++r)
        Cs[w][lhi * 8 + r][ni * 16 + lrow] = acc[ch][ni][r];
    __syncthreads();
#pragma unroll
    for (int rr = 0; rr < 8; ++rr) {
      int row = r0 + rr * 2;               // 0..15 within chunk
      float4 g4 = *(const float4*)&Cs[w][row][h * 4];
      int gm = m0 + wm * 64 + ch * 16 + row;
      int hg = hg0 + h;
      float cv = cell[(size_t)gm * HID + hg];
      float pf = g4.x + b4.x, pi = g4.y + b4.y, po = g4.z + b4.z, pc = g4.w + b4.w;
      float f = 1.0f / (1.0f + __expf(-pf));
      float i = 1.0f / (1.0f + __expf(-pi));
      float o = 1.0f / (1.0f + __expf(-po));
      float e2 = __expf(-2.0f * pc);
      float c = (1.0f - e2) / (1.0f + e2);         // tanh
      float nc = f * cv + i * c;
      float e2c = __expf(-2.0f * nc);
      float th = (1.0f - e2c) / (1.0f + e2c);      // tanh
      out_h[(size_t)gm * HID + hg] = o * th;
      out_c[(size_t)gm * HID + hg] = nc;
    }
  }
}

extern "C" void kernel_launch(void* const* d_in, const int* in_sizes, int n_in,
                              void* d_out, int out_size, void* d_ws, size_t ws_size,
                              hipStream_t stream) {
  const float* x      = (const float*)d_in[0];
  const float* hidden = (const float*)d_in[1];
  const float* cell   = (const float*)d_in[2];
  const float* Wx[4] = {(const float*)d_in[3], (const float*)d_in[5],
                        (const float*)d_in[7], (const float*)d_in[9]};
  const float* bx[4] = {(const float*)d_in[4], (const float*)d_in[6],
                        (const float*)d_in[8], (const float*)d_in[10]};
  const float* Wh[4] = {(const float*)d_in[11], (const float*)d_in[13],
                        (const float*)d_in[15], (const float*)d_in[17]};
  const float* bh[4] = {(const float*)d_in[12], (const float*)d_in[14],
                        (const float*)d_in[16], (const float*)d_in[18]};

  unsigned short* Abf = (unsigned short*)d_ws;                                     // 16 MB
  unsigned short* Bbf = (unsigned short*)((char*)d_ws + ((size_t)16 << 20));       // 16 MB
  float*          bia = (float*)((char*)d_ws + ((size_t)32 << 20));                // 16 KB

  float* out_h = (float*)d_out;
  float* out_c = out_h + (size_t)BATCH * HID;

  pack_a_kernel<<<(BATCH * KDIM) / (256 * 8), 256, 0, stream>>>(x, hidden, Abf);
  pack_b_kernel<<<NDIM, 256, 0, stream>>>(Wx[0], Wx[1], Wx[2], Wx[3],
                                          Wh[0], Wh[1], Wh[2], Wh[3],
                                          bx[0], bx[1], bx[2], bx[3],
                                          bh[0], bh[1], bh[2], bh[3], Bbf, bia);
  dim3 grid(NDIM / BN, BATCH / BM);
  lstm_gemm_kernel<<<grid, 128, 0, stream>>>(Abf, Bbf, bia, cell, out_h, out_c);
}
